// IDGCN_86139864089095
// MI455X (gfx1250) — matrix operand order repaired and checked by reading, not compiled, not run on hardware
//
#include <hip/hip_runtime.h>
#include <math.h>

// ---------------------------------------------------------------------------
// IDGCN forward for MI455X (gfx1250): bf16 WMMA 16x16x32 for all GEMM-shaped
// work (temporal convs, graph mix, 1x1 channel mix), fp32 accumulate.
// All WMMA operands staged in LDS so each lane's fragment is contiguous
// 16B runs -> ds_load_b128 / ds_store_b128, no per-element packing.
// B=8, C=64, N=170, T=512 (half T = 256).
// ---------------------------------------------------------------------------

#define Bb 8
#define Cc 64
#define Nn 170
#define Th 256
#define SZ (Bb*Cc*Nn*Th)       // 22,282,240 elements per big buffer
#define NN2 (Nn*Nn)            // 28,900

typedef __attribute__((ext_vector_type(16))) __bf16 bf16x16;
typedef __attribute__((ext_vector_type(8)))  __bf16 bf16x8;
typedef __attribute__((ext_vector_type(8)))  float  f32x8;

// ---- WMMA fragment helpers (CDNA5 wave32 layouts, CDNA5 ISA) ----

// A: 16x32 bf16 tile from row-major [M][K] (leading dim ld, multiple of 8).
// lane: row m = m0 + (lane&15); half = lane>>4;
// elements 0..7  -> K = k0 + 8*half + e      (16B contiguous)
// elements 8..15 -> K = k0 + 16 + 8*half + e (16B contiguous)
__device__ __forceinline__ bf16x16 load_a(const __bf16* W, int ld, int m0, int k0) {
  int lane = threadIdx.x & 31;
  const __bf16* row = W + (m0 + (lane & 15)) * ld + k0 + ((lane >> 4) << 3);
  bf16x8 lo = *(const bf16x8*)row;
  bf16x8 hi = *(const bf16x8*)(row + 16);
  return __builtin_shufflevector(lo, hi, 0,1,2,3,4,5,6,7,8,9,10,11,12,13,14,15);
}

// B: 32x16 bf16 tile from TRANSPOSED LDS [N][K] (leading dim ld, multiple of 8).
// lane: n = n0 + (lane&15); half = lane>>4;
// elements 0..7 -> K = k0 + 8*half + e, elements 8..15 -> K = k0 + 16 + 8*half + e
__device__ __forceinline__ bf16x16 load_b_t(const __bf16* XT, int ld, int k0, int n0) {
  int lane = threadIdx.x & 31;
  const __bf16* p = XT + (n0 + (lane & 15)) * ld + k0 + ((lane >> 4) << 3);
  bf16x8 lo = *(const bf16x8*)p;
  bf16x8 hi = *(const bf16x8*)(p + 16);
  return __builtin_shufflevector(lo, hi, 0,1,2,3,4,5,6,7,8,9,10,11,12,13,14,15);
}

__device__ __forceinline__ f32x8 wmma_bf16(bf16x16 a, bf16x16 b, f32x8 c) {
  return __builtin_amdgcn_wmma_f32_16x16x32_bf16(false, a, false, b, (short)0, c,
                                                 false, false);
}

__device__ __forceinline__ float gumbel_noise(unsigned seed, int r, int m) {
  unsigned h = seed * 0x9E3779B9u ^ (unsigned)(r * 0x85EBCA6B) ^ (unsigned)(m * 0xC2B2AE35);
  h ^= h >> 16; h *= 0x7FEB352Du; h ^= h >> 15; h *= 0x846CA68Bu; h ^= h >> 16;
  float u = (float)(h >> 8) * (1.0f / 16777216.0f) + 1e-10f;
  return -logf(-logf(u + 1e-20f) + 1e-20f);
}

// ---------------------------------------------------------------------------
// adj = softmax(relu(nv1 @ nv2), axis=1).  One block per row n.
// ---------------------------------------------------------------------------
__global__ __launch_bounds__(256)
void k_adj(const float* __restrict__ nv1, const float* __restrict__ nv2,
           float* __restrict__ adj) {
  int n = blockIdx.x, tid = threadIdx.x;
  __shared__ float z[176];
  __shared__ float red[256];
  float v = -3.0e38f;
  if (tid < Nn) {
    float acc = 0.f;
#pragma unroll
    for (int k = 0; k < 10; ++k) acc += nv1[n*10 + k] * nv2[k*Nn + tid];
    acc = acc > 0.f ? acc : 0.f;
    z[tid] = acc; v = acc;
  }
  red[tid] = v; __syncthreads();
  for (int s = 128; s > 0; s >>= 1) { if (tid < s) red[tid] = fmaxf(red[tid], red[tid+s]); __syncthreads(); }
  float mx = red[0]; __syncthreads();
  float e = 0.f;
  if (tid < Nn) { e = __expf(z[tid] - mx); z[tid] = e; }
  red[tid] = e; __syncthreads();
  for (int s = 128; s > 0; s >>= 1) { if (tid < s) red[tid] += red[tid+s]; __syncthreads(); }
  float sum = red[0];
  if (tid < Nn) adj[n*Nn + tid] = z[tid] / sum;
}

// ---------------------------------------------------------------------------
// Deinterleave x (T=512) into xe/xo (T=256).
// ---------------------------------------------------------------------------
__global__ __launch_bounds__(256)
void k_deint(const float* __restrict__ x, float* __restrict__ xe, float* __restrict__ xo) {
  int i = blockIdx.x * 256 + threadIdx.x;
  int t = i & 255, rest = i >> 8;
  int src = (rest << 9) + (t << 1);
  xe[i] = x[src];
  xo[i] = x[src + 1];
}

// ---------------------------------------------------------------------------
// Fused branch: tanh(conv3(leaky(conv5(edge_pad(x))))) via WMMA.
// One block per (b, n, T-tile of 64). B operands transposed in LDS.
// LDS: 40960 + 12096 + 9504 = 62560 B.
// ---------------------------------------------------------------------------
__global__ __launch_bounds__(256)
void k_branch(const float* __restrict__ xin,
              const float* __restrict__ w5, const float* __restrict__ b5,
              const float* __restrict__ w3, const float* __restrict__ b3,
              float* __restrict__ hout) {
  __shared__ __align__(16) __bf16 wbuf[5*64*64];   // [tap][o][c], w5 then w3
  __shared__ __align__(16) __bf16 xsT[84*72];      // [u][c]  u = t-3..t+80 (clamped)
  __shared__ __align__(16) __bf16 ysT[66*72];      // [tp][o] conv5+leaky intermediate
  const int tid = threadIdx.x, wave = tid >> 5;
  const int t0 = blockIdx.x * 64, n = blockIdx.y, b = blockIdx.z;

  // w5 (coalesced flat read, scatter to [k][o][c])
  for (int i = tid; i < 5*64*64; i += 256) {
    int o = i / 320, rem = i % 320, c = rem / 5, k = rem % 5;
    wbuf[k*4096 + o*64 + c] = (__bf16)w5[i];
  }
  // input tile, transposed: xsT[u][c]
  for (int i = tid; i < 64*84; i += 256) {
    int c = i / 84, u = i % 84;
    int t = t0 + u - 3; t = t < 0 ? 0 : (t > 255 ? 255 : t);
    xsT[u*72 + c] = (__bf16)xin[(((b*64 + c)*Nn) + n)*Th + t];
  }
  __syncthreads();

  // Phase 1: y5[o, tp], tp in [0,66): 4 o-tiles x 5 tp-tiles, 5 taps, K=64.
  for (int tile = wave; tile < 20; tile += 8) {
    int mt = tile / 5, pt = tile % 5;
    f32x8 acc = {0.f,0.f,0.f,0.f,0.f,0.f,0.f,0.f};
#pragma unroll
    for (int k = 0; k < 5; ++k) {
#pragma unroll
      for (int kc = 0; kc < 64; kc += 32) {
        bf16x16 af = load_a(&wbuf[k*4096], 64, mt*16, kc);
        bf16x16 bf = load_b_t(xsT, 72, kc, pt*16 + k);
        acc = wmma_bf16(af, bf, acc);
      }
    }
    int lane = tid & 31, half = lane >> 4, col = lane & 15;
    int tp = pt*16 + col;
    bf16x8 pk;
#pragma unroll
    for (int r = 0; r < 8; ++r) {
      int o = mt*16 + r + 8*half;
      float v = acc[r] + b5[o];
      v = v >= 0.f ? v : 0.01f * v;                 // leaky relu
      pk[r] = (__bf16)v;
    }
    if (tp < 66)  // D-layout: 8 consecutive M per lane -> one ds_store_b128
      *(bf16x8*)&ysT[tp*72 + mt*16 + 8*half] = pk;
  }
  __syncthreads();
  for (int i = tid; i < 3*64*64; i += 256) {
    int o = i / 192, rem = i % 192, c = rem / 3, k = rem % 3;
    wbuf[k*4096 + o*64 + c] = (__bf16)w3[i];
  }
  __syncthreads();

  // Phase 2: h[o, t] = tanh(conv3(y5)): 4 o-tiles x 4 t-tiles, 3 taps, K=64.
  for (int tile = wave; tile < 16; tile += 8) {
    int mt = tile >> 2, nt = tile & 3;
    f32x8 acc = {0.f,0.f,0.f,0.f,0.f,0.f,0.f,0.f};
#pragma unroll
    for (int k = 0; k < 3; ++k) {
#pragma unroll
      for (int kc = 0; kc < 64; kc += 32) {
        bf16x16 af = load_a(&wbuf[k*4096], 64, mt*16, kc);
        bf16x16 bf = load_b_t(ysT, 72, kc, nt*16 + k);
        acc = wmma_bf16(af, bf, acc);
      }
    }
    int lane = tid & 31, half = lane >> 4, col = lane & 15;
    int t = t0 + nt*16 + col;
#pragma unroll
    for (int r = 0; r < 8; ++r) {
      int o = mt*16 + r + 8*half;
      hout[(((b*64 + o)*Nn) + n)*Th + t] = tanhf(acc[r] + b3[o]);
    }
  }
}

// ---------------------------------------------------------------------------
// s1[c][n] = sum_t h[7,c,n,t]  (only batch -1 feeds graph_gen)
// ---------------------------------------------------------------------------
__global__ __launch_bounds__(256)
void k_rsum(const float* __restrict__ h, float* __restrict__ s1) {
  int n = blockIdx.x, c = blockIdx.y, tid = threadIdx.x;
  __shared__ float red[256];
  red[tid] = h[(((7*64 + c)*Nn) + n)*Th + tid];
  __syncthreads();
  for (int s = 128; s > 0; s >>= 1) { if (tid < s) red[tid] += red[tid+s]; __syncthreads(); }
  if (tid == 0) s1[c*Nn + n] = red[0];
}

// ---------------------------------------------------------------------------
// Collapsed graph_gen + dadj mix. One block per row n of the output adjacency.
// Forward value of the gumbel straight-through is exactly the one-hot of
// argmax(t2 + g) (row-constant logsumexp shift dropped).
// ---------------------------------------------------------------------------
__global__ __launch_bounds__(256)
void k_ggen(const float* __restrict__ s1, const float* __restrict__ adj,
            const float* __restrict__ ggw, const float* __restrict__ ggb,
            const float* __restrict__ fc0w, const float* __restrict__ fc0b,
            const float* __restrict__ fc1w, const float* __restrict__ fc1b,
            const float* __restrict__ fc2w, const float* __restrict__ fc2b,
            const float* __restrict__ a, unsigned seed,
            float* __restrict__ dadj) {
  int r = blockIdx.x, tid = threadIdx.x;
  __shared__ float s1c[64], s2c[64], hrow[64], t0r[170], t1r[340], t2r[170];
  __shared__ float rv[256]; __shared__ int ri[256];

  if (tid < 64) {
    int c = tid;
    s1c[c] = s1[c*Nn + r];
    float acc = 0.f;
    for (int m = 0; m < Nn; ++m) acc += s1[c*Nn + m] * adj[m*Nn + r];
    s2c[c] = acc;
  }
  __syncthreads();
  if (tid < 64) {
    int o = tid; float acc = ggb[o] * 256.0f;
    for (int c = 0; c < 64; ++c)
      acc += ggw[o*128 + c] * s1c[c] + ggw[o*128 + 64 + c] * s2c[c];
    hrow[o] = acc;
  }
  __syncthreads();
  if (tid < Nn) {
    float acc = fc0b[tid];
    for (int o = 0; o < 64; ++o) acc += hrow[o] * fc0w[tid*64 + o];
    t0r[tid] = tanhf(acc);
  }
  __syncthreads();
  for (int j = tid; j < 2*Nn; j += 256) {
    float acc = fc1b[j];
    for (int i2 = 0; i2 < Nn; ++i2) acc += t0r[i2] * fc1w[j*Nn + i2];
    t1r[j] = tanhf(acc);
  }
  __syncthreads();
  if (tid < Nn) {
    float acc = fc2b[tid];
    for (int i2 = 0; i2 < 2*Nn; ++i2) acc += t1r[i2] * fc2w[tid*2*Nn + i2];
    t2r[tid] = tanhf(acc);
  }
  __syncthreads();
  float sc = -3.0e38f;
  if (tid < Nn) sc = t2r[tid] + gumbel_noise(seed, r, tid);
  rv[tid] = sc; ri[tid] = tid;
  __syncthreads();
  for (int s = 128; s > 0; s >>= 1) {
    if (tid < s && rv[tid + s] > rv[tid]) { rv[tid] = rv[tid+s]; ri[tid] = ri[tid+s]; }
    __syncthreads();
  }
  int amax = ri[0];
  float av = a[0];
  if (tid < Nn) {
    float la = (tid == amax && tid != r) ? 1.0f : 0.0f;
    dadj[r*Nn + tid] = la * av + adj[r*Nn + tid] * (1.0f - av);
  }
}

// ---------------------------------------------------------------------------
// gm[b,c,m,t] = sum_n h[b,c,n,t] * dadj[n,m]  via WMMA (M=m, K=n pad->192, N=t)
// grid.x encodes (t-tile 0..3, m-group 0..2 of 64 rows), grid.y=c, grid.z=b.
// LDS: at 25600 + hsT 25600 = 51200 B. Zero-padding lives on the B side so
// any A padding garbage multiplies 0.
// ---------------------------------------------------------------------------
__global__ __launch_bounds__(256)
void k_gmix(const float* __restrict__ h, const float* __restrict__ dadj,
            float* __restrict__ gm) {
  __shared__ __align__(16) __bf16 at[64*200];   // at[m_local][n] = dadj[n][m]
  __shared__ __align__(16) __bf16 hsT[64*200];  // hsT[tl][n], n >= 170 zeroed
  const int tid = threadIdx.x, wave = tid >> 5;
  const int tt = blockIdx.x & 3, mg = blockIdx.x >> 2;
  const int c = blockIdx.y, b = blockIdx.z;
  const int t0 = tt * 64;

  for (int i = tid; i < 64*192; i += 256) {       // coalesced over m
    int nn = i >> 6, ml = i & 63;
    int m = mg*64 + ml;
    at[ml*200 + nn] = (__bf16)((m < Nn && nn < Nn) ? dadj[nn*Nn + m] : 0.f);
  }
  for (int i = tid; i < 64*192; i += 256) {       // coalesced over t
    int nn = i >> 6, tl = i & 63;
    hsT[tl*200 + nn] =
        (__bf16)(nn < Nn ? h[(((b*64 + c)*Nn) + nn)*Th + t0 + tl] : 0.f);
  }
  __syncthreads();

  for (int tile = wave; tile < 16; tile += 8) {
    int mt = tile >> 2, nt = tile & 3;
    f32x8 acc = {0.f,0.f,0.f,0.f,0.f,0.f,0.f,0.f};
#pragma unroll
    for (int kc = 0; kc < 192; kc += 32) {
      bf16x16 af = load_a(at, 200, mt*16, kc);
      bf16x16 bf = load_b_t(hsT, 200, kc, nt*16);
      acc = wmma_bf16(af, bf, acc);
    }
    int lane = tid & 31, half = lane >> 4, col = lane & 15;
    int t = t0 + nt*16 + col;
#pragma unroll
    for (int r = 0; r < 8; ++r) {
      int m = mg*64 + mt*16 + r + 8*half;
      if (m < Nn) gm[(((b*64 + c)*Nn) + m)*Th + t] = acc[r];
    }
  }
}

// ---------------------------------------------------------------------------
// so = h + dcw[:, :64] @ h + dcw[:, 64:] @ gm + dcb  (fused 1x1 + residual)
// LDS: wl 16384 + xkT 17408 = 33792 B.
// ---------------------------------------------------------------------------
__global__ __launch_bounds__(256)
void k_cmix(const float* __restrict__ h, const float* __restrict__ gm,
            const float* __restrict__ dcw, const float* __restrict__ dcb,
            float* __restrict__ so) {
  __shared__ __align__(16) __bf16 wl[64*128];   // [o][k]
  __shared__ __align__(16) __bf16 xkT[64*136];  // [tl][k] rows k: 0..63 h, 64..127 gm
  const int tid = threadIdx.x, wave = tid >> 5;
  const int t0 = blockIdx.x * 64, n = blockIdx.y, b = blockIdx.z;

  for (int i = tid; i < 64*128; i += 256) wl[i] = (__bf16)dcw[i];
  for (int i = tid; i < 128*64; i += 256) {
    int k = i >> 6, tl = i & 63;                  // coalesced over t
    float v = (k < 64) ? h[(((b*64 + k)*Nn) + n)*Th + t0 + tl]
                       : gm[(((b*64 + (k-64))*Nn) + n)*Th + t0 + tl];
    xkT[tl*136 + k] = (__bf16)v;
  }
  __syncthreads();

  for (int tile = wave; tile < 16; tile += 8) {
    int mt = tile >> 2, nt = tile & 3;
    f32x8 acc = {0.f,0.f,0.f,0.f,0.f,0.f,0.f,0.f};
#pragma unroll
    for (int kc = 0; kc < 128; kc += 32) {
      bf16x16 af = load_a(wl, 128, mt*16, kc);
      bf16x16 bf = load_b_t(xkT, 136, kc, nt*16);
      acc = wmma_bf16(af, bf, acc);
    }
    int lane = tid & 31, half = lane >> 4, col = lane & 15;
    int t = t0 + nt*16 + col;
#pragma unroll
    for (int r = 0; r < 8; ++r) {
      int o = mt*16 + r + 8*half;
      int idx = (((b*64 + o)*Nn) + n)*Th + t;
      so[idx] = acc[r] + dcb[o] + h[idx];   // fp32 residual from global
    }
  }
}

// ---------------------------------------------------------------------------
// Elementwise glue
// ---------------------------------------------------------------------------
__global__ __launch_bounds__(256)
void k_multanh(const float* __restrict__ a, const float* __restrict__ b,
               float* __restrict__ out) {
  int i = blockIdx.x * 256 + threadIdx.x;
  out[i] = a[i] * tanhf(b[i]);
}
__global__ __launch_bounds__(256)
void k_sub(const float* __restrict__ a, const float* __restrict__ b,
           float* __restrict__ out) {
  int i = blockIdx.x * 256 + threadIdx.x;
  out[i] = a[i] - b[i];
}
__global__ __launch_bounds__(256)
void k_add(const float* __restrict__ a, const float* __restrict__ b,
           float* __restrict__ out) {
  int i = blockIdx.x * 256 + threadIdx.x;
  out[i] = a[i] + b[i];
}
__global__ __launch_bounds__(256)
void k_copy(const float* __restrict__ a, float* __restrict__ out, int n) {
  int i = blockIdx.x * 256 + threadIdx.x;
  if (i < n) out[i] = a[i];
}

// ---------------------------------------------------------------------------
// Host side
// ---------------------------------------------------------------------------
static void run_step(const float* in,
                     const float* w5, const float* b5,
                     const float* w3, const float* b3,
                     const float* ggw, const float* ggb,
                     const float* fc0w, const float* fc0b,
                     const float* fc1w, const float* fc1b,
                     const float* fc2w, const float* fc2b,
                     const float* dcw, const float* dcb,
                     const float* a, const float* adj,
                     float* hbuf, float* gm, float* so,
                     float* s1, float* dadj, unsigned seed, hipStream_t stream) {
  k_branch<<<dim3(4, Nn, Bb), 256, 0, stream>>>(in, w5, b5, w3, b3, hbuf);
  k_rsum<<<dim3(Nn, Cc), 256, 0, stream>>>(hbuf, s1);
  k_ggen<<<Nn, 256, 0, stream>>>(s1, adj, ggw, ggb, fc0w, fc0b, fc1w, fc1b,
                                 fc2w, fc2b, a, seed, dadj);
  k_gmix<<<dim3(12, Cc, Bb), 256, 0, stream>>>(hbuf, dadj, gm);
  k_cmix<<<dim3(4, Nn, Bb), 256, 0, stream>>>(hbuf, gm, dcw, dcb, so);
}

extern "C" void kernel_launch(void* const* d_in, const int* in_sizes, int n_in,
                              void* d_out, int out_size, void* d_ws, size_t ws_size,
                              hipStream_t stream) {
  const float* x    = (const float*)d_in[0];
  const float* nv1  = (const float*)d_in[1];
  const float* nv2  = (const float*)d_in[2];
  const float* a    = (const float*)d_in[3];
  const float* cw5[4] = {(const float*)d_in[4],  (const float*)d_in[8],
                         (const float*)d_in[12], (const float*)d_in[16]};
  const float* cb5[4] = {(const float*)d_in[5],  (const float*)d_in[9],
                         (const float*)d_in[13], (const float*)d_in[17]};
  const float* cw3[4] = {(const float*)d_in[6],  (const float*)d_in[10],
                         (const float*)d_in[14], (const float*)d_in[18]};
  const float* cb3[4] = {(const float*)d_in[7],  (const float*)d_in[11],
                         (const float*)d_in[15], (const float*)d_in[19]};
  const float* ggw  = (const float*)d_in[20];
  const float* ggb  = (const float*)d_in[21];
  const float* fc0w = (const float*)d_in[22];
  const float* fc0b = (const float*)d_in[23];
  const float* fc1w = (const float*)d_in[24];
  const float* fc1b = (const float*)d_in[25];
  const float* fc2w = (const float*)d_in[26];
  const float* fc2b = (const float*)d_in[27];
  const float* dcw  = (const float*)d_in[28];
  const float* dcb  = (const float*)d_in[29];

  float* ws   = (float*)d_ws;
  float* xe   = ws + 0*(size_t)SZ;
  float* xo   = ws + 1*(size_t)SZ;
  float* hbuf = ws + 2*(size_t)SZ;
  float* gm   = ws + 3*(size_t)SZ;
  float* so   = ws + 4*(size_t)SZ;
  float* dbuf = ws + 5*(size_t)SZ;
  float* adj  = ws + 6*(size_t)SZ;
  float* dadj = adj + NN2;
  float* s1   = dadj + NN2;
  float* out  = (float*)d_out;

  const int EW = SZ / 256;  // elementwise grid

  k_adj<<<Nn, 256, 0, stream>>>(nv1, nv2, adj);
  k_deint<<<EW, 256, 0, stream>>>(x, xe, xo);

  // step 1: x1 = step(xe); d = xo * tanh(x1)
  run_step(xe, cw5[0], cb5[0], cw3[0], cb3[0], ggw, ggb, fc0w, fc0b,
           fc1w, fc1b, fc2w, fc2b, dcw, dcb, a, adj,
           hbuf, gm, so, s1, dadj, 0x51u, stream);
  k_multanh<<<EW, 256, 0, stream>>>(xo, so, dbuf);

  // step 2: x2 = step(xo); c = xe * tanh(x2)  (c overwrites xe in place)
  run_step(xo, cw5[1], cb5[1], cw3[1], cb3[1], ggw, ggb, fc0w, fc0b,
           fc1w, fc1b, fc2w, fc2b, dcw, dcb, a, adj,
           hbuf, gm, so, s1, dadj, 0x52u, stream);
  k_multanh<<<EW, 256, 0, stream>>>(xe, so, xe);

  // step 3: x3 = step(c); x_odd = d - x3 -> d_out[SZ:2SZ]
  run_step(xe, cw5[2], cb5[2], cw3[2], cb3[2], ggw, ggb, fc0w, fc0b,
           fc1w, fc1b, fc2w, fc2b, dcw, dcb, a, adj,
           hbuf, gm, so, s1, dadj, 0x53u, stream);
  k_sub<<<EW, 256, 0, stream>>>(dbuf, so, out + (size_t)SZ);

  // step 4: x4 = step(d); x_even = c + x4 -> d_out[0:SZ]; dadj -> tail
  run_step(dbuf, cw5[3], cb5[3], cw3[3], cb3[3], ggw, ggb, fc0w, fc0b,
           fc1w, fc1b, fc2w, fc2b, dcw, dcb, a, adj,
           hbuf, gm, so, s1, dadj, 0x54u, stream);
  k_add<<<EW, 256, 0, stream>>>(xe, so, out);
  k_copy<<<(NN2 + 255) / 256, 256, 0, stream>>>(dadj, out + 2*(size_t)SZ, NN2);
}
